// SimpleGCN_16466904613348
// MI455X (gfx1250) — matrix operand order repaired and checked by reading, not compiled, not run on hardware
//
#include <hip/hip_runtime.h>
#include <hip/hip_bf16.h>
#include <cstdint>

// ---------------- problem constants (match reference) ----------------
#define NN   40000      // nodes
#define EE   1280000    // edges
#define INF  500        // in_feats
#define KPAD 512        // K padded for WMMA (k=32 steps)
#define FF   8          // fields
#define HD   128        // hidden
#define CC   41         // classes
#define PP   28         // triu pairs of 8 fields

// ---------------- CDNA5 WMMA types ----------------
typedef __attribute__((ext_vector_type(16))) __bf16 v16bf;
typedef __attribute__((ext_vector_type(8)))  float  v8f;
typedef __attribute__((ext_vector_type(4)))  unsigned int v4u;
typedef __attribute__((ext_vector_type(4)))  float  v4f;
typedef int vsi4 __attribute__((vector_size(16)));   // matches builtin param type

union Frag {                 // one 16x32 (A) or 32x16 (B) bf16 fragment, 8 VGPRs
    unsigned short u[16];
    v4u q[2];
    v16bf v;
};

__device__ __forceinline__ unsigned short f2bf(float f) {
    unsigned int u = __float_as_uint(f);
    u += 0x7FFFu + ((u >> 16) & 1u);       // round-to-nearest-even
    return (unsigned short)(u >> 16);
}

// ---------------- CDNA5 async global->LDS DMA (ASYNCcnt path) ----------------
#if __has_builtin(__builtin_amdgcn_global_load_async_to_lds_b128)
#define ASYNC_LDS 1
#else
#define ASYNC_LDS 0
#endif

__device__ __forceinline__ void async_g2l_b128(const unsigned short* g, unsigned short* l) {
#if ASYNC_LDS
    __builtin_amdgcn_global_load_async_to_lds_b128(
        (__attribute__((address_space(1))) vsi4*)g,
        (__attribute__((address_space(3))) vsi4*)l,
        0, 0);
#else
    *(v4u*)l = *(const v4u*)g;             // fallback: VGPR bounce copy
#endif
}

__device__ __forceinline__ void wait_async_all() {
#if ASYNC_LDS
#if __has_builtin(__builtin_amdgcn_s_wait_asynccnt)
    __builtin_amdgcn_s_wait_asynccnt(0);
#else
    asm volatile("s_wait_asynccnt 0" ::: "memory");
#endif
#endif
}

__device__ __constant__ int ROWc[PP] = {0,0,0,0,0,0,0,1,1,1,1,1,1,2,2,2,2,2,3,3,3,3,4,4,4,5,5,6};
__device__ __constant__ int COLc[PP] = {1,2,3,4,5,6,7,2,3,4,5,6,7,3,4,5,6,7,4,5,6,7,5,6,7,6,7,7};

// ============== stage 1: degree / norm ==============
__global__ void deg_kernel(const int* __restrict__ dst, float* __restrict__ deg) {
    int e = blockIdx.x * blockDim.x + threadIdx.x;
    if (e < EE) atomicAdd(&deg[dst[e]], 1.0f);
}

__global__ void norm_kernel(float* __restrict__ deg) {
    int i = blockIdx.x * blockDim.x + threadIdx.x;
    if (i < NN) { float d = deg[i]; deg[i] = rsqrtf(d > 0.f ? d : 1.0f); }
}

// ============== stage 2: one propagation hop (gather*norm[src] -> atomic scatter) ==============
__global__ void hop_scatter(const float* __restrict__ hin, const float* __restrict__ norm,
                            const int* __restrict__ src, const int* __restrict__ dst,
                            float* __restrict__ hout, long long total /* EE*125 */) {
    long long t = (long long)blockIdx.x * blockDim.x + threadIdx.x;
    if (t >= total) return;
    int e  = (int)(t / 125);
    int c4 = (int)(t % 125) * 4;
    int s = src[e], d = dst[e];
    float ns = norm[s];
    v4f v = *(const v4f*)(hin + (long long)s * INF + c4);
    float* o = hout + (long long)d * INF + c4;
    atomicAdd(o + 0, v[0] * ns);
    atomicAdd(o + 1, v[1] * ns);
    atomicAdd(o + 2, v[2] * ns);
    atomicAdd(o + 3, v[3] * ns);
}

__global__ void scale_rows(float* __restrict__ buf, const float* __restrict__ norm,
                           long long total /* NN*INF */) {
    long long t = (long long)blockIdx.x * blockDim.x + threadIdx.x;
    if (t >= total) return;
    buf[t] *= norm[t / INF];
}

// ============== stage 3: weight conversion to bf16 ==============
__global__ void conv_wsg(const float* __restrict__ W, unsigned short* __restrict__ O) {
    int t = blockIdx.x * blockDim.x + threadIdx.x;          // KPAD*HD
    if (t >= KPAD * HD) return;
    int r = t >> 7;
    O[t] = (r < INF) ? f2bf(W[r * HD + (t & 127)]) : (unsigned short)0;
}

__global__ void conv_watt(const float* __restrict__ W, unsigned short* __restrict__ O) {
    int t = blockIdx.x * blockDim.x + threadIdx.x;          // HD*HD
    if (t < HD * HD) O[t] = f2bf(W[t]);
}

// ============== stage 4: h = prop2 @ W_sg + b_sg via bf16 WMMA ==============
// 4 waves/block; each wave owns a 16(node) x 128 tile. B tiles (32x128 bf16, 8KB)
// are shared block-wide, staged into LDS by async DMA, double buffered so the
// DMA of tile k+1 overlaps the 8 WMMAs of tile k.
__global__ __launch_bounds__(128) void sg_gemm(const float* __restrict__ A,
                                               const unsigned short* __restrict__ Wbf,
                                               const float* __restrict__ bias,
                                               float* __restrict__ Hout) {
    __shared__ unsigned short Bt[2][32 * HD];               // 2 x 8KB

    int wid  = threadIdx.x >> 5;
    int lane = threadIdx.x & 31;
    int gw   = blockIdx.x * 4 + wid;                        // NN/16 = 2500 waves exactly
    int m0   = gw * 16;
    int hs = lane >> 4, mr = lane & 15;
    const float* arow = A + (long long)(m0 + mr) * INF;

    // stage B tile kt into LDS buffer buf (512 x b128 chunks, 4 per thread)
    auto stage = [&](int kt, int buf) {
        const unsigned short* gsrc = Wbf + (long long)kt * 32 * HD;
        unsigned short* ldst = Bt[buf];
#pragma unroll
        for (int i = 0; i < 4; ++i) {
            int idx = threadIdx.x + i * 128;                // 0..511
            async_g2l_b128(gsrc + idx * 8, ldst + idx * 8);
        }
    };

    v8f acc[8];
#pragma unroll
    for (int nt = 0; nt < 8; ++nt) {
        float b = bias[nt * 16 + mr];
#pragma unroll
        for (int r = 0; r < 8; ++r) acc[nt][r] = b;
    }

    stage(0, 0);
    wait_async_all();
    __syncthreads();

    for (int kt = 0; kt < KPAD / 32; ++kt) {
        int buf = kt & 1;
        if (kt < KPAD / 32 - 1) stage(kt + 1, buf ^ 1);     // overlap DMA with compute

        int kb = kt * 32;
        // A fragment: lane<16 -> K {kb..kb+7, kb+16..kb+23}; lane>=16 -> +8
        Frag af;
        const float* p0 = arow + kb + 8 * hs;
        const float* p1 = arow + kb + 16 + 8 * hs;
        v4f x0 = *(const v4f*)p0, x1 = *(const v4f*)(p0 + 4);
        v4f y0 = *(const v4f*)p1, y1 = *(const v4f*)(p1 + 4);
#pragma unroll
        for (int i = 0; i < 4; ++i) { af.u[i]     = f2bf(x0[i]); af.u[4 + i]  = f2bf(x1[i]); }
#pragma unroll
        for (int i = 0; i < 4; ++i) { af.u[8 + i] = f2bf(y0[i]); af.u[12 + i] = f2bf(y1[i]); }

        const unsigned short* brow = Bt[buf] + mr;          // B col N = mr, K {8hs..8hs+7, 16+8hs..16+8hs+7}
#pragma unroll
        for (int nt = 0; nt < 8; ++nt) {
            Frag bf_;
#pragma unroll
            for (int i = 0; i < 8; ++i) {
                bf_.u[i]     = brow[(8 * hs + i) * HD + nt * 16];
                bf_.u[8 + i] = brow[(16 + 8 * hs + i) * HD + nt * 16];
            }
            acc[nt] = __builtin_amdgcn_wmma_f32_16x16x32_bf16(
                false, af.v, false, bf_.v, (short)0, acc[nt], false, false);
        }
        wait_async_all();
        __syncthreads();                                    // next buffer ready, old consumed
    }
#pragma unroll
    for (int nt = 0; nt < 8; ++nt)
#pragma unroll
        for (int r = 0; r < 8; ++r)
            Hout[(long long)(m0 + r + 8 * hs) * HD + nt * 16 + mr] = acc[nt][r];
}

// ============== stage 5: fused AFM + final linear ==============
// 2 waves/block, 16 nodes/block: W_att (32KB bf16) is async-DMA'd into LDS once
// per block and reused for all 16 nodes, so B fragments come from LDS.
__global__ __launch_bounds__(64) void afm_kernel(const int* __restrict__ nzidx,
                                                 const float* __restrict__ nzval,
                                                 const float* __restrict__ emb,
                                                 const unsigned short* __restrict__ WattBf,
                                                 const float* __restrict__ b_att,
                                                 const float* __restrict__ Hfeat,
                                                 const float* __restrict__ W_fin,
                                                 const float* __restrict__ b_fin,
                                                 float* __restrict__ out) {
    __shared__ unsigned short watt_lds[HD * HD];            // 32KB
    __shared__ float          fv_s[2][FF * HD];             // field embeddings (f32)
    __shared__ unsigned short ip_s[2][32 * HD];             // ip as bf16, M padded to 32
    __shared__ float          hl_s[2][HD];
    __shared__ float          sc_s[2][32];
    __shared__ float          at_s[2][HD];

    int wid = threadIdx.x >> 5, lane = threadIdx.x & 31;
    int hs = lane >> 4, mr = lane & 15;

    // --- stage W_att into LDS (2048 b128 chunks over 64 threads) ---
#pragma unroll
    for (int i = 0; i < 32; ++i) {
        int idx = threadIdx.x + i * 64;                     // 0..2047
        async_g2l_b128(WattBf + idx * 8, watt_lds + idx * 8);
    }
    wait_async_all();
    __syncthreads();

    float*          fvw = fv_s[wid];
    unsigned short* ipw = ip_s[wid];

    for (int it = 0; it < 8; ++it) {
        int node = blockIdx.x * 16 + it * 2 + wid;          // NN = 2500*16 exactly

        // --- fv[f][c] = emb[idx[f]][c] * val[f]; stage h into LDS ---
#pragma unroll
        for (int f = 0; f < FF; ++f) {
            int idx = nzidx[node * FF + f];
            float val = nzval[node * FF + f];
            const float* erow = emb + (long long)idx * HD;
#pragma unroll
            for (int i = 0; i < 4; ++i) {
                int c = lane + 32 * i;
                fvw[f * HD + c] = erow[c] * val;
            }
        }
#pragma unroll
        for (int i = 0; i < 4; ++i) {
            int c = lane + 32 * i;
            hl_s[wid][c] = Hfeat[(long long)node * HD + c];
        }

        // --- ip (bf16) for WMMA A, rows 28..31 zero ---
#pragma unroll
        for (int p = 0; p < PP; ++p) {
            int r = ROWc[p], q = COLc[p];
#pragma unroll
            for (int i = 0; i < 4; ++i) {
                int c = lane + 32 * i;
                ipw[p * HD + c] = f2bf(fvw[r * HD + c] * fvw[q * HD + c]);
            }
        }
#pragma unroll
        for (int p = PP; p < 32; ++p)
#pragma unroll
            for (int i = 0; i < 4; ++i) ipw[p * HD + lane + 32 * i] = 0;
        __syncthreads();

        // --- preload all 8 A fragments (2 m-tiles x 4 k-tiles) from LDS ---
        Frag af[2][4];
#pragma unroll
        for (int mt = 0; mt < 2; ++mt) {
            const unsigned short* base = ipw + (mt * 16 + mr) * HD;
#pragma unroll
            for (int kt = 0; kt < 4; ++kt) {
                af[mt][kt].q[0] = *(const v4u*)(base + kt * 32 + 8 * hs);
                af[mt][kt].q[1] = *(const v4u*)(base + kt * 32 + 16 + 8 * hs);
            }
        }

        // --- fm = relu(ip @ W_att + b_att); fold into score dot with h ---
        float s0[8], s1[8];
#pragma unroll
        for (int r = 0; r < 8; ++r) { s0[r] = 0.f; s1[r] = 0.f; }

#pragma unroll
        for (int nt = 0; nt < 8; ++nt) {
            float batt = b_att[nt * 16 + mr];
            v8f a0, a1;
#pragma unroll
            for (int r = 0; r < 8; ++r) { a0[r] = batt; a1[r] = batt; }
#pragma unroll
            for (int kt = 0; kt < 4; ++kt) {
                Frag bf_;
                const unsigned short* bp = watt_lds + (kt * 32) * HD + nt * 16 + mr;
#pragma unroll
                for (int i = 0; i < 8; ++i) {
                    bf_.u[i]     = bp[(8 * hs + i) * HD];
                    bf_.u[8 + i] = bp[(16 + 8 * hs + i) * HD];
                }
                a0 = __builtin_amdgcn_wmma_f32_16x16x32_bf16(false, af[0][kt].v, false, bf_.v,
                                                             (short)0, a0, false, false);
                a1 = __builtin_amdgcn_wmma_f32_16x16x32_bf16(false, af[1][kt].v, false, bf_.v,
                                                             (short)0, a1, false, false);
            }
            float hc = hl_s[wid][nt * 16 + mr];             // D col of this lane
#pragma unroll
            for (int r = 0; r < 8; ++r) {
                s0[r] += fmaxf(a0[r], 0.f) * hc;
                s1[r] += fmaxf(a1[r], 0.f) * hc;
            }
        }

        // --- reduce scores across the 16 lanes holding each row ---
#pragma unroll
        for (int r = 0; r < 8; ++r) {
            float a = s0[r], b = s1[r];
            for (int m = 1; m <= 8; m <<= 1) {
                a += __shfl_xor(a, m, 32);
                b += __shfl_xor(b, m, 32);
            }
            if (mr == r) {
                sc_s[wid][r + 8 * hs]      = a;             // rows 0..15  (m-tile 0)
                sc_s[wid][16 + r + 8 * hs] = b;             // rows 16..31 (m-tile 1)
            }
        }
        __syncthreads();

        // --- softmax over 28 pairs (wave-wide shuffles) ---
        float sv = (lane < PP) ? sc_s[wid][lane] : -1e30f;
        float mx = sv;
        for (int m = 16; m >= 1; m >>= 1) mx = fmaxf(mx, __shfl_xor(mx, m, 32));
        float e = expf(sv - mx);
        if (lane >= PP) e = 0.f;
        float sum = e;
        for (int m = 16; m >= 1; m >>= 1) sum += __shfl_xor(sum, m, 32);
        sc_s[wid][lane] = e / sum;
        __syncthreads();

        // --- attn_out = 100 * sum_p score[p] * ip[p][:] (recompute from fv, f32) ---
#pragma unroll
        for (int i = 0; i < 4; ++i) {
            int c = lane + 32 * i;
            float a = 0.f;
#pragma unroll
            for (int p = 0; p < PP; ++p)
                a += sc_s[wid][p] * fvw[ROWc[p] * HD + c] * fvw[COLc[p] * HD + c];
            at_s[wid][c] = a * 100.f;
        }
        __syncthreads();

        // --- final: out = [h | attn] @ W_fin + b_fin  (256x41 GEMV per node) ---
        for (int j = lane; j < CC; j += 32) {
            float o = b_fin[j];
#pragma unroll 4
            for (int k = 0; k < HD; ++k) {
                o += hl_s[wid][k] * W_fin[k * CC + j];
                o += at_s[wid][k] * W_fin[(HD + k) * CC + j];
            }
            out[(long long)node * CC + j] = o;
        }
        __syncthreads();                                    // buffers reused next iteration
    }
}

// ============================== host launcher ==============================
extern "C" void kernel_launch(void* const* d_in, const int* in_sizes, int n_in,
                              void* d_out, int out_size, void* d_ws, size_t ws_size,
                              hipStream_t stream) {
    const float* x     = (const float*)d_in[0];
    const float* nzval = (const float*)d_in[1];
    const float* emb   = (const float*)d_in[2];
    const float* W_sg  = (const float*)d_in[3];
    const float* b_sg  = (const float*)d_in[4];
    const float* W_att = (const float*)d_in[5];
    const float* b_att = (const float*)d_in[6];
    const float* W_fin = (const float*)d_in[7];
    const float* b_fin = (const float*)d_in[8];
    const int*   eidx  = (const int*)d_in[9];
    const int*   nzidx = (const int*)d_in[10];
    const int* src = eidx;
    const int* dst = eidx + EE;

    // workspace layout (all 256B aligned)
    char* ws = (char*)d_ws;
    size_t off = 0;
    auto take = [&](size_t bytes) { char* p = ws + off; off = (off + bytes + 255) & ~(size_t)255; return p; };
    float*          norm  = (float*)take((size_t)NN * 4);
    float*          tmp1  = (float*)take(((size_t)NN * INF + 64) * 4);
    float*          tmp2  = (float*)take(((size_t)NN * INF + 64) * 4);
    float*          h     = (float*)take((size_t)NN * HD * 4);
    unsigned short* wsgb  = (unsigned short*)take((size_t)KPAD * HD * 2);
    unsigned short* wattb = (unsigned short*)take((size_t)HD * HD * 2);
    (void)ws_size; (void)in_sizes; (void)n_in; (void)out_size;

    (void)hipMemsetAsync(norm, 0, (size_t)NN * 4, stream);
    (void)hipMemsetAsync(tmp1, 0, ((size_t)NN * INF + 64) * 4, stream);
    (void)hipMemsetAsync(tmp2, 0, ((size_t)NN * INF + 64) * 4, stream);

    // degree + norm
    deg_kernel<<<(EE + 255) / 256, 256, 0, stream>>>(dst, norm);
    norm_kernel<<<(NN + 255) / 256, 256, 0, stream>>>(norm);

    // bf16 weight staging (independent; overlaps with propagation)
    conv_wsg<<<(KPAD * HD + 255) / 256, 256, 0, stream>>>(W_sg, wsgb);
    conv_watt<<<(HD * HD + 255) / 256, 256, 0, stream>>>(W_att, wattb);

    // hop 1: x -> tmp1
    const long long hop_total = (long long)EE * (INF / 4);     // 160M float4 tasks
    const long long row_total = (long long)NN * INF;
    hop_scatter<<<(unsigned)((hop_total + 255) / 256), 256, 0, stream>>>(x, norm, src, dst, tmp1, hop_total);
    scale_rows<<<(unsigned)((row_total + 255) / 256), 256, 0, stream>>>(tmp1, norm, row_total);

    // hop 2: tmp1 -> tmp2
    hop_scatter<<<(unsigned)((hop_total + 255) / 256), 256, 0, stream>>>(tmp1, norm, src, dst, tmp2, hop_total);
    scale_rows<<<(unsigned)((row_total + 255) / 256), 256, 0, stream>>>(tmp2, norm, row_total);

    // h = tmp2 @ W_sg + b_sg   (WMMA bf16, async-LDS double-buffered B)
    sg_gemm<<<NN / 16 / 4, 128, 0, stream>>>(tmp2, wsgb, b_sg, h);

    // fused AFM + final linear (WMMA bf16, W_att staged in LDS, 16 nodes/block)
    afm_kernel<<<NN / 16, 64, 0, stream>>>(nzidx, nzval, emb, wattb, b_att, h,
                                           W_fin, b_fin, (float*)d_out);
}
